// Head_90177133346839
// MI455X (gfx1250) — matrix operand order repaired and checked by reading, not compiled, not run on hardware
//
#include <hip/hip_runtime.h>
#include <hip/hip_bf16.h>

// ---------------------------------------------------------------------------
// Shapes (fixed by the reference): x[B,T,D], W*[D,D], b*[D]; out spk[B,T,D] f32
// ---------------------------------------------------------------------------
#define B_DIM 4
#define T_DIM 4096
#define D_DIM 1024
#define M_DIM (B_DIM * T_DIM)   // 16384 GEMM rows

typedef __attribute__((ext_vector_type(16))) __bf16 v16bf;
typedef __attribute__((ext_vector_type(8)))  __bf16 v8bf;
typedef __attribute__((ext_vector_type(8)))  float  v8f;

// ---------------------------------------------------------------------------
// Kernel 0: streaming f32 -> bf16 conversion (one pass; x and the 3 weights).
// 8 elements per thread, vectorized: global_load_b128 x2 -> cvt_pk -> b128 st.
// ---------------------------------------------------------------------------
__global__ __launch_bounds__(256)
void cvt_f32_to_bf16(const float* __restrict__ in, __bf16* __restrict__ out)
{
    const size_t i = ((size_t)blockIdx.x * blockDim.x + threadIdx.x) * 8;
    v8f v = *(const v8f*)(in + i);
    v8bf o;
#pragma unroll
    for (int j = 0; j < 8; ++j) o[j] = (__bf16)v[j];
    *(v8bf*)(out + i) = o;
}

// ---------------------------------------------------------------------------
// Kernel 1: fused triple projection GEMM (bf16 WMMA, f32 accum) + activations.
//   out[m,n] = sum_k x[m,k] * W[n,k]   (einsum "btd,ed->bte" == x @ W^T)
//
// Workgroup: 256 threads = 8 waves arranged 4(row) x 2(col).
// Each wave computes TWO 16-row tiles x one 16-col tile per output matrix:
//   block tile = 128 rows x 32 cols, 6 accumulators/wave,
//   6 WMMAs per K-step sharing 3 B fragments and 2 A fragments.
//
// Fragment k order used here (contiguous, same for A and B; not the ISA table order):
//   A: lane l (m = l&15, h = l>>4) holds K = {16h..16h+7} in a[0..7] and
//      K = {16h+8..16h+15} in a[8..15]  -> two contiguous 16B bf16 chunks.
//   B: lane l (n = l&15, h = l>>4) holds K = {16h..16h+15} of column n,
//      and B[k][n] = W[n][k]            -> one contiguous 32B chunk of W row.
// ---------------------------------------------------------------------------
__global__ __launch_bounds__(256)
void proj_gemm_bf16_wmma(const __bf16* __restrict__ xb,
                         const __bf16* __restrict__ Wr, const float* __restrict__ br,
                         const __bf16* __restrict__ Wk, const float* __restrict__ bk,
                         const __bf16* __restrict__ Wv, const float* __restrict__ bv,
                         __bf16* __restrict__ r_out,
                         __bf16* __restrict__ ek_out,
                         __bf16* __restrict__ v_out)
{
    const int K = D_DIM;
    const int N = D_DIM;

    const int wave = threadIdx.x >> 5;
    const int lane = threadIdx.x & 31;
    const int wrow = wave & 3;            // 0..3  (32-row slabs)
    const int wcol = wave >> 2;           // 0..1  (16-col slabs)
    const int half = lane >> 4;           // 0/1
    const int ml   = lane & 15;

    const int rowBase = blockIdx.y * 128 + wrow * 32;   // two 16-row tiles
    const int colBase = blockIdx.x * 32  + wcol * 16;

    const __bf16* __restrict__ xrow0 = xb + (size_t)(rowBase + ml) * K + 16 * half;
    const __bf16* __restrict__ xrow1 = xrow0 + (size_t)16 * K;
    const __bf16* __restrict__ wrR   = Wr + (size_t)(colBase + ml) * K + 16 * half;
    const __bf16* __restrict__ wkR   = Wk + (size_t)(colBase + ml) * K + 16 * half;
    const __bf16* __restrict__ wvR   = Wv + (size_t)(colBase + ml) * K + 16 * half;

    v8f cr0 = {}, ck0 = {}, cv0 = {};
    v8f cr1 = {}, ck1 = {}, cv1 = {};

    for (int kk = 0; kk < K; kk += 32) {
        // ---- A fragments: two contiguous 8xbf16 (16B) chunks each ----
        v8bf a0lo = *(const v8bf*)(xrow0 + kk);
        v8bf a0hi = *(const v8bf*)(xrow0 + kk + 8);
        v8bf a1lo = *(const v8bf*)(xrow1 + kk);
        v8bf a1hi = *(const v8bf*)(xrow1 + kk + 8);
        v16bf af0 = __builtin_shufflevector(a0lo, a0hi,
                        0,1,2,3,4,5,6,7,8,9,10,11,12,13,14,15);
        v16bf af1 = __builtin_shufflevector(a1lo, a1hi,
                        0,1,2,3,4,5,6,7,8,9,10,11,12,13,14,15);

        // ---- B fragments: one contiguous 16xbf16 (32B) chunk each ----
        v16bf bfr = *(const v16bf*)(wrR + kk);
        v16bf bfk = *(const v16bf*)(wkR + kk);
        v16bf bfv = *(const v16bf*)(wvR + kk);

        // prefetch next K-slab of the x streams (global_prefetch_b8)
        __builtin_prefetch(xrow0 + kk + 32, 0, 0);
        __builtin_prefetch(xrow1 + kk + 32, 0, 0);

        // ---- 6 WMMAs: B fragments reused across both row tiles ----
        cr0 = __builtin_amdgcn_wmma_f32_16x16x32_bf16(false, af0, false, bfr,
                                                      (short)0, cr0, false, false);
        cr1 = __builtin_amdgcn_wmma_f32_16x16x32_bf16(false, af1, false, bfr,
                                                      (short)0, cr1, false, false);
        ck0 = __builtin_amdgcn_wmma_f32_16x16x32_bf16(false, af0, false, bfk,
                                                      (short)0, ck0, false, false);
        ck1 = __builtin_amdgcn_wmma_f32_16x16x32_bf16(false, af1, false, bfk,
                                                      (short)0, ck1, false, false);
        cv0 = __builtin_amdgcn_wmma_f32_16x16x32_bf16(false, af0, false, bfv,
                                                      (short)0, cv0, false, false);
        cv1 = __builtin_amdgcn_wmma_f32_16x16x32_bf16(false, af1, false, bfv,
                                                      (short)0, cv1, false, false);
    }

    // ---- epilogue: bias + activation, store bf16 streams --------------------
    // C/D layout: lane (n = lane&15, h = lane>>4); VGPR j holds M = j + 8h.
    const int n  = colBase + ml;
    const float brn = br[n];
    const float bkn = bk[n];
    const float bvn = bv[n];
#pragma unroll
    for (int j = 0; j < 8; ++j) {
        const int m0 = rowBase + j + 8 * half;
        const size_t off0 = (size_t)m0 * N + n;
        const size_t off1 = off0 + (size_t)16 * N;

        r_out[off0]  = (__bf16)(1.0f / (1.0f + __expf(-(cr0[j] + brn))));
        ek_out[off0] = (__bf16)__expf(ck0[j] + bkn);
        v_out[off0]  = (__bf16)(cv0[j] + bvn);

        r_out[off1]  = (__bf16)(1.0f / (1.0f + __expf(-(cr1[j] + brn))));
        ek_out[off1] = (__bf16)__expf(ck1[j] + bkn);
        v_out[off1]  = (__bf16)(cv1[j] + bvn);
    }
}

// ---------------------------------------------------------------------------
// Kernel 2: sequential WKV + LIF scan. One thread per (b,d) channel.
// Loads of r/ek/v at each t are coalesced (consecutive d across the wave).
// ---------------------------------------------------------------------------
__global__ __launch_bounds__(256)
void wkv_lif_scan(const __bf16* __restrict__ r_s,
                  const __bf16* __restrict__ ek_s,
                  const __bf16* __restrict__ v_s,
                  const float* __restrict__ b_td,
                  const float* __restrict__ gdecay,
                  float* __restrict__ out)
{
    const int d = blockIdx.x * blockDim.x + threadIdx.x;   // 0..D-1
    const int b = blockIdx.y;                              // 0..B-1

    const float td = __expf(b_td[d]);
    const float g  = __expf(gdecay[d]);

    const size_t base = (size_t)b * T_DIM * D_DIM + d;
    const __bf16* pr = r_s  + base;
    const __bf16* pk = ek_s + base;
    const __bf16* pv = v_s  + base;
    float*       po  = out  + base;

    float sa = 0.0f, sb = 0.0f, mem = 0.0f;

    for (int t = 0; t < T_DIM; ++t) {
        // stream a few steps ahead (address-independent of the recurrence)
        __builtin_prefetch(pk + 8 * D_DIM, 0, 0);

        const float ekt = (float)*pk;
        const float vt  = (float)*pv;
        const float rt  = (float)*pr;

        sa = ekt * vt + td * sa;
        sb = ekt + td * sb;
        const float wkv = (g * ekt * vt + sa) / (g * ekt + sb + 1e-6f);
        const float inp = rt * wkv;

        const float reset = (mem - 0.05f) > 0.0f ? 1.0f : 0.0f;  // prev-mem spike
        mem = 0.3f * mem + inp - reset * 0.05f;
        *po = (mem - 0.05f) > 0.0f ? 1.0f : 0.0f;

        pr += D_DIM; pk += D_DIM; pv += D_DIM; po += D_DIM;
    }
}

// ---------------------------------------------------------------------------
// launch
// ---------------------------------------------------------------------------
extern "C" void kernel_launch(void* const* d_in, const int* in_sizes, int n_in,
                              void* d_out, int out_size, void* d_ws, size_t ws_size,
                              hipStream_t stream) {
    const float* x      = (const float*)d_in[0];
    const float* W_r    = (const float*)d_in[1];
    const float* b_r    = (const float*)d_in[2];
    const float* W_k    = (const float*)d_in[3];
    const float* b_k    = (const float*)d_in[4];
    const float* W_v    = (const float*)d_in[5];
    const float* b_v    = (const float*)d_in[6];
    // d_in[7] = W_td: provably dead (TokenShift input is all-zero)
    const float* b_td   = (const float*)d_in[8];
    const float* gdecay = (const float*)d_in[9];
    float* out = (float*)d_out;

    const size_t MD = (size_t)M_DIM * D_DIM;    // 16M elements
    const size_t DD = (size_t)D_DIM * D_DIM;    // 1M elements

    // workspace layout (bf16): x (32MB) | Wr|Wk|Wv (3x2MB) | r|ek|v (3x32MB)
    __bf16* xb    = (__bf16*)d_ws;
    __bf16* wrb   = xb   + MD;
    __bf16* wkb   = wrb  + DD;
    __bf16* wvb   = wkb  + DD;
    __bf16* r_ws  = wvb  + DD;
    __bf16* ek_ws = r_ws + MD;
    __bf16* v_ws  = ek_ws + MD;

    // Pass 0: one-time f32 -> bf16 conversion of GEMM operands
    cvt_f32_to_bf16<<<dim3((unsigned)(MD / 8 / 256)), 256, 0, stream>>>(x,   xb);
    cvt_f32_to_bf16<<<dim3((unsigned)(DD / 8 / 256)), 256, 0, stream>>>(W_r, wrb);
    cvt_f32_to_bf16<<<dim3((unsigned)(DD / 8 / 256)), 256, 0, stream>>>(W_k, wkb);
    cvt_f32_to_bf16<<<dim3((unsigned)(DD / 8 / 256)), 256, 0, stream>>>(W_v, wvb);

    // Pass 1: GEMM, tiles of 128 rows x 32 cols
    dim3 ggrid(D_DIM / 32, M_DIM / 128);        // (32, 128)
    proj_gemm_bf16_wmma<<<ggrid, 256, 0, stream>>>(
        xb, wrb, b_r, wkb, b_k, wvb, b_v, r_ws, ek_ws, v_ws);

    // Pass 2: scan, one thread per (b,d) channel
    dim3 sgrid(D_DIM / 256, B_DIM);             // (4, 4)
    wkv_lif_scan<<<sgrid, 256, 0, stream>>>(r_ws, ek_ws, v_ws, b_td, gdecay, out);
}
